// Attn_PatchLevel_46050639348312
// MI455X (gfx1250) — matrix operand order repaired and checked by reading, not compiled, not run on hardware
//
#include <hip/hip_runtime.h>
#include <hip/hip_bf16.h>
#include <stdint.h>

// ---------------------------------------------------------------------------
// Types for CDNA5 WMMA (gfx1250, wave32)
// ---------------------------------------------------------------------------
typedef __attribute__((ext_vector_type(16))) __bf16 v16bf;
typedef __attribute__((ext_vector_type(8)))  float  v8f;

union BfPack { uint4 u[2]; v16bf v; };

// round-to-nearest-even f32 -> bf16 (raw u16)
__device__ __forceinline__ unsigned short f2bf(float f) {
    union { float f; unsigned u; } v; v.f = f;
    unsigned r = v.u + 0x7FFFu + ((v.u >> 16) & 1u);
    return (unsigned short)(r >> 16);
}

// ---------------------------------------------------------------------------
// CDNA5 async global->LDS copy (ASYNCcnt-tracked, no VGPR round trip).
// Low 32 bits of a generic pointer into LDS == LDS byte offset.
// ---------------------------------------------------------------------------
__device__ __forceinline__ void async_g2l_b128(void* ldsDst, const void* gSrc) {
    unsigned lofs = (unsigned)(uintptr_t)ldsDst;
    asm volatile("global_load_async_to_lds_b128 %0, %1, off"
                 :: "v"(lofs), "v"(gSrc)
                 : "memory");
}
__device__ __forceinline__ void wait_asynccnt0() {
    asm volatile("s_wait_asynccnt 0x0" ::: "memory");
}

// Force schedule: one batch of DS-reads, then the WMMA chain.
// masks: 0x100 = DS read, 0x008 = MFMA/WMMA
template <int NDS, int NWMMA>
__device__ __forceinline__ void sched_batch() {
    __builtin_amdgcn_sched_group_barrier(0x100, NDS, 0);
    __builtin_amdgcn_sched_group_barrier(0x008, NWMMA, 0);
}

// ---------------------------------------------------------------------------
// Operand loaders (LDS, row-major u16 tiles)
//
// A operand (16x32, M x K): lane L -> M = L%16 ; VGPR v<4 -> K = 2v + (L<16?0:8),
//                           VGPR v>=4 -> K = 16 + 2(v-4) + (L<16?0:8)
// B operand (32x16, K x N): lane L -> N = L%16 ; same K mapping as A
// ---------------------------------------------------------------------------
__device__ __forceinline__ v16bf load_A(const unsigned short* lds, int row0,
                                        int ld, int k0, int m16, int hi) {
    BfPack p;
    const uint4* q = (const uint4*)(lds + (size_t)(row0 + m16) * ld + k0 + hi * 8);
    p.u[0] = q[0];   // K = k0 + hi*8 .. +7
    p.u[1] = q[2];   // K = k0 + 16 + hi*8 .. +7
    return p.v;
}

__device__ __forceinline__ v16bf load_B(const unsigned short* lds, int n0,
                                        int ld, int k0, int m16, int hi) {
    BfPack p;
    const uint4* q = (const uint4*)(lds + (size_t)(n0 + m16) * ld + k0 + hi * 8);
    p.u[0] = q[0];
    p.u[1] = q[2];
    return p.v;
}

__device__ __forceinline__ v8f wmma_bf16(v16bf a, v16bf b, v8f c) {
    return __builtin_amdgcn_wmma_f32_16x16x32_bf16(false, a, false, b,
                                                   (short)0, c, false, false);
}

// ---------------------------------------------------------------------------
// Kernel 1: convert the 3 weight matrices (each 512x512 f32) to bf16 in ws.
// ---------------------------------------------------------------------------
__global__ void cvt_weights_kernel(const float* __restrict__ Wq,
                                   const float* __restrict__ Wkv,
                                   const float* __restrict__ Wo,
                                   unsigned short* __restrict__ out) {
    int i = blockIdx.x * 256 + threadIdx.x;
    if (i < 262144)       out[i] = f2bf(Wq[i]);
    else if (i < 524288)  out[i] = f2bf(Wkv[i - 262144]);
    else if (i < 786432)  out[i] = f2bf(Wo[i - 524288]);
}

// ---------------------------------------------------------------------------
// Kernel 2: Y[M,512] = cvt_bf16(X_f32[M,512]) @ W_bf16[512,512]^T + bias
// Block: 256 threads (8 waves), 128-row output tile.
// W tiles double-buffered in LDS via async global->LDS copies overlapped with
// WMMA work on the other buffer (one barrier per tile). Per k-step the
// schedule is pinned as two half-batches: (A + 4xB DS-reads, 4 WMMAs) x 2 —
// counted dscnt waits without exceeding the VGPR budget.
// LDS 166KB => 1 block/WGP; __launch_bounds__(256,1) frees the full VGPR file.
// LDS: X 128x520 bf16 (133120 B) | W[2] 128x72 bf16 (2x18432 B) = 169984 B
// ---------------------------------------------------------------------------
#define GEMM_LDS_BYTES (133120 + 2 * 18432)

__global__ void __launch_bounds__(256, 1)
gemm_bias_kernel(const float* __restrict__ X,
                 const unsigned short* __restrict__ Wb,
                 const float* __restrict__ bias,
                 unsigned short* __restrict__ outB,
                 float* __restrict__ outF) {
    extern __shared__ char sm[];
    unsigned short* ldsX = (unsigned short*)sm;   // [128][520]
    char* wbase = sm + 133120;                    // two [128][72] buffers
    const int K = 512, N = 512;

    const int row0 = blockIdx.x * 128;
    const int tid  = threadIdx.x;
    const int w    = tid >> 5;
    const int lane = tid & 31;
    const int m16  = lane & 15;
    const int hi   = lane >> 4;
    const int strip = w * 16;

    // Prime the pipeline: async-stage W tile 0 (n0=0, kc=0) into buffer 0.
    {
        unsigned short* wdst = (unsigned short*)wbase;
        for (int i = tid; i < 128 * 8; i += 256) {
            int r = i >> 3, c = i & 7;
            async_g2l_b128(wdst + (size_t)r * 72 + c * 8,
                           Wb + (size_t)r * K + c * 8);
        }
    }

    // Stage X tile (f32 -> bf16): 128 rows x 512 cols (overlaps with async W)
    for (int i = tid; i < 128 * 128; i += 256) {
        int r = i >> 7, c4 = i & 127;
        float4 f = ((const float4*)(X + (size_t)(row0 + r) * K))[c4];
        unsigned short* d = ldsX + (size_t)r * 520 + c4 * 4;
        d[0] = f2bf(f.x); d[1] = f2bf(f.y); d[2] = f2bf(f.z); d[3] = f2bf(f.w);
    }

    v8f acc[8];
    int cur = 0;
    // 32 tiles: j = n-chunk*8 + k-chunk
    for (int j = 0; j < 32; j++) {
        const int n0 = (j >> 3) << 7;   // 0,128,256,384
        const int kc = (j & 7) << 6;    // 0..448

        wait_asynccnt0();     // our async copies (tile j) done
        __syncthreads();      // everyone's copies done + prior reads finished

        // Issue async stage of tile j+1 into the other buffer.
        if (j + 1 < 32) {
            const int n1 = ((j + 1) >> 3) << 7;
            const int k1 = ((j + 1) & 7) << 6;
            unsigned short* wdst = (unsigned short*)(wbase + (cur ^ 1) * 18432);
            for (int i = tid; i < 128 * 8; i += 256) {
                int r = i >> 3, c = i & 7;
                async_g2l_b128(wdst + (size_t)r * 72 + c * 8,
                               Wb + (size_t)(n1 + r) * K + k1 + c * 8);
            }
        }

        if ((j & 7) == 0) {
#pragma unroll
            for (int t = 0; t < 8; t++) { v8f z = {}; acc[t] = z; }
        }

        const unsigned short* wbuf = (const unsigned short*)(wbase + cur * 18432);
#pragma unroll
        for (int ks = 0; ks < 64; ks += 32) {
            v16bf a = load_A(ldsX, strip, 520, kc + ks, m16, hi);
            v16bf b[4];
#pragma unroll
            for (int nt = 0; nt < 4; nt++)
                b[nt] = load_B(wbuf, nt * 16, 72, ks, m16, hi);
#pragma unroll
            for (int nt = 0; nt < 4; nt++)
                acc[nt] = wmma_bf16(a, b[nt], acc[nt]);
            sched_batch<10, 4>();
#pragma unroll
            for (int nt = 0; nt < 4; nt++)
                b[nt] = load_B(wbuf, (nt + 4) * 16, 72, ks, m16, hi);
#pragma unroll
            for (int nt = 0; nt < 4; nt++)
                acc[nt + 4] = wmma_bf16(a, b[nt], acc[nt + 4]);
            sched_batch<8, 4>();
        }

        if ((j & 7) == 7) {
            // Store: C/D layout -> lane L: N = L%16, VGPR r: M = strip + r + hi*8
#pragma unroll
            for (int nt = 0; nt < 8; nt++) {
                int n = n0 + nt * 16 + m16;
                float bv = bias[n];
#pragma unroll
                for (int r = 0; r < 8; r++) {
                    int row = row0 + strip + r + hi * 8;
                    float v = acc[nt][r] + bv;
                    if (outF) outF[(size_t)row * N + n] = v;
                    else      outB[(size_t)row * N + n] = f2bf(v);
                }
            }
        }
        cur ^= 1;
    }
}

// ---------------------------------------------------------------------------
// Kernel 3: per-head attention.
// grid = (P/64, B*V), block = 128 threads (4 waves). 64 query rows per block.
//   phase 1: scores = scale * Q Kᵀ with diagonal -inf mask  -> LDS f32
//            (4 s-tile accumulators at once: one K sweep, A reused by 4 WMMAs)
//   phase 2: softmax (one thread per row)                    -> bf16 over Q buf
//   phase 3: out = P V, V staged transposed in LDS           -> f32 workspace
// Q and K tiles staged with async global->LDS copies; DS/WMMA schedule pinned.
// LDS 260KB => 1 block/WGP; __launch_bounds__(128,1).
// LDS: [Q/P 64x520 bf16 | scores 64x516 f32 == VT 128x520 bf16 | K 64x520 bf16]
// ---------------------------------------------------------------------------
#define ATTN_LDS_BYTES (66560 + 133120 + 66560)

__global__ void __launch_bounds__(128, 1)
attention_kernel(const unsigned short* __restrict__ Qp,
                 const unsigned short* __restrict__ Kp,
                 const unsigned short* __restrict__ Vp,
                 float* __restrict__ attnOut) {
    extern __shared__ char sm[];
    unsigned short* ldsQP = (unsigned short*)sm;              // 64 x 520 bf16
    float*          ldsS  = (float*)(sm + 66560);             // 64 x 516 f32
    unsigned short* ldsVT = (unsigned short*)(sm + 66560);    // 128 x 520 bf16
    unsigned short* ldsK  = (unsigned short*)(sm + 199680);   // 64 x 520 bf16

    const int P = 512, D = 512;
    const int bh = blockIdx.y;
    const int q0 = blockIdx.x * 64;
    const size_t base = (size_t)bh * P * D;

    const int tid  = threadIdx.x;  // 128
    const int w    = tid >> 5;
    const int lane = tid & 31;
    const int m16  = lane & 15;
    const int hi   = lane >> 4;
    const int strip = w * 16;

    const float scale = 0.044194173824159216f;  // 1/sqrt(512)

    // Async-stage Q tile (64 x 512 bf16); completion folded into first K wait.
    for (int i = tid; i < 64 * 64; i += 128) {
        int r = i >> 6, c = i & 63;
        async_g2l_b128(ldsQP + (size_t)r * 520 + c * 8,
                       Qp + base + (size_t)(q0 + r) * D + c * 8);
    }

    // ---- phase 1: scores -------------------------------------------------
    for (int s0 = 0; s0 < P; s0 += 64) {
        __syncthreads();  // previous ldsK consumers done
        for (int i = tid; i < 64 * 64; i += 128) {
            int r = i >> 6, c = i & 63;
            async_g2l_b128(ldsK + (size_t)r * 520 + c * 8,
                           Kp + base + (size_t)(s0 + r) * D + c * 8);
        }
        wait_asynccnt0();
        __syncthreads();

        v8f acc4[4];
#pragma unroll
        for (int t = 0; t < 4; t++) { v8f z = {}; acc4[t] = z; }

        for (int k0 = 0; k0 < D; k0 += 32) {
            v16bf a = load_A(ldsQP, strip, 520, k0, m16, hi);
            v16bf b[4];
#pragma unroll
            for (int st = 0; st < 4; st++)
                b[st] = load_B(ldsK, st * 16, 520, k0, m16, hi);
#pragma unroll
            for (int st = 0; st < 4; st++)
                acc4[st] = wmma_bf16(a, b[st], acc4[st]);
            sched_batch<10, 4>();
        }

#pragma unroll
        for (int st = 0; st < 4; st++) {
            int s = s0 + st * 16 + m16;
#pragma unroll
            for (int r = 0; r < 8; r++) {
                int mrow = strip + r + hi * 8;
                float v = acc4[st][r] * scale;
                if (s == q0 + mrow) v = -__builtin_inff();  // OffDiag mask
                ldsS[(size_t)mrow * 516 + s] = v;
            }
        }
    }
    __syncthreads();

    // ---- phase 2: softmax (threads 0..63, one row each) ------------------
    if (tid < 64) {
        float* row = ldsS + (size_t)tid * 516;
        float m = -__builtin_inff();
        for (int s = 0; s < P; s++) m = fmaxf(m, row[s]);
        float sum = 0.f;
        for (int s = 0; s < P; s++) {
            float e = __expf(row[s] - m);
            row[s] = e;
            sum += e;
        }
        float inv = 1.0f / sum;
        unsigned short* prow = ldsQP + (size_t)tid * 520;  // overwrite Q buf
        for (int s = 0; s < P; s++) prow[s] = f2bf(row[s] * inv);
    }

    // ---- phase 3: out = P @ V -------------------------------------------
    for (int n0 = 0; n0 < D; n0 += 128) {
        __syncthreads();  // scores/VT consumers done
        // Stage V chunk transposed: ldsVT[d_local][s] = V[s][n0 + d_local]
        for (int i = tid; i < 512 * 32; i += 128) {
            int s = i >> 5, c4 = i & 31;
            uint2 v = ((const uint2*)(Vp + base + (size_t)s * D + n0))[c4];
            int d = c4 * 4;
            ldsVT[(size_t)(d + 0) * 520 + s] = (unsigned short)(v.x & 0xFFFFu);
            ldsVT[(size_t)(d + 1) * 520 + s] = (unsigned short)(v.x >> 16);
            ldsVT[(size_t)(d + 2) * 520 + s] = (unsigned short)(v.y & 0xFFFFu);
            ldsVT[(size_t)(d + 3) * 520 + s] = (unsigned short)(v.y >> 16);
        }
        __syncthreads();

        v8f acc[8];
#pragma unroll
        for (int t = 0; t < 8; t++) { v8f z = {}; acc[t] = z; }

        for (int k0 = 0; k0 < P; k0 += 32) {
            v16bf a = load_A(ldsQP, strip, 520, k0, m16, hi);  // P matrix
            v16bf b[4];
#pragma unroll
            for (int nt = 0; nt < 4; nt++)
                b[nt] = load_B(ldsVT, nt * 16, 520, k0, m16, hi);
#pragma unroll
            for (int nt = 0; nt < 4; nt++)
                acc[nt] = wmma_bf16(a, b[nt], acc[nt]);
            sched_batch<10, 4>();
#pragma unroll
            for (int nt = 0; nt < 4; nt++)
                b[nt] = load_B(ldsVT, (nt + 4) * 16, 520, k0, m16, hi);
#pragma unroll
            for (int nt = 0; nt < 4; nt++)
                acc[nt + 4] = wmma_bf16(a, b[nt], acc[nt + 4]);
            sched_batch<8, 4>();
        }

#pragma unroll
        for (int nt = 0; nt < 8; nt++) {
            int n = n0 + nt * 16 + m16;
#pragma unroll
            for (int r = 0; r < 8; r++) {
                int mrow = strip + r + hi * 8;
                attnOut[base + (size_t)(q0 + mrow) * D + n] = acc[nt][r];
            }
        }
    }
}

// ---------------------------------------------------------------------------
// Host-side launcher
// Inputs: 0 queries, 1 keys, 2 values, 3 Wq, 4 bq, 5 Wkv, 6 bkv, 7 Wo, 8 bo
// Workspace layout (bytes):
//   [0, 1572864)            : Wq|Wkv|Wo as bf16
//   [1572864, +67108864)    : q projected (bf16)
//   next 67108864           : k projected (bf16)
//   next 67108864           : v projected (bf16)
//   next 134217728          : attention output (f32)
// ---------------------------------------------------------------------------
extern "C" void kernel_launch(void* const* d_in, const int* in_sizes, int n_in,
                              void* d_out, int out_size, void* d_ws, size_t ws_size,
                              hipStream_t stream) {
    const float* queries = (const float*)d_in[0];
    const float* keys    = (const float*)d_in[1];
    const float* values  = (const float*)d_in[2];
    const float* Wq  = (const float*)d_in[3];
    const float* bq  = (const float*)d_in[4];
    const float* Wkv = (const float*)d_in[5];
    const float* bkv = (const float*)d_in[6];
    const float* Wo  = (const float*)d_in[7];
    const float* bo  = (const float*)d_in[8];

    char* ws = (char*)d_ws;
    unsigned short* wsW  = (unsigned short*)ws;
    unsigned short* qp   = (unsigned short*)(ws + 1572864);
    unsigned short* kp   = (unsigned short*)(ws + 1572864 + 67108864);
    unsigned short* vp   = (unsigned short*)(ws + 1572864 + 2 * 67108864);
    float*          attn = (float*)(ws + 1572864 + (size_t)3 * 67108864);

    // 1) weights f32 -> bf16
    cvt_weights_kernel<<<3072, 256, 0, stream>>>(Wq, Wkv, Wo, wsW);

    // 2) projections (M = B*V*P = 65536 rows each)
    gemm_bias_kernel<<<512, 256, GEMM_LDS_BYTES, stream>>>(
        queries, wsW, bq, qp, nullptr);
    gemm_bias_kernel<<<512, 256, GEMM_LDS_BYTES, stream>>>(
        keys, wsW + 262144, bkv, kp, nullptr);
    gemm_bias_kernel<<<512, 256, GEMM_LDS_BYTES, stream>>>(
        values, wsW + 262144, bkv, vp, nullptr);

    // 3) attention: grid (P/64 = 8, B*V = 128)
    attention_kernel<<<dim3(8, 128), 128, ATTN_LDS_BYTES, stream>>>(
        qp, kp, vp, attn);

    // 4) output projection -> f32 d_out
    gemm_bias_kernel<<<512, 256, GEMM_LDS_BYTES, stream>>>(
        attn, wsW + 524288, bo, nullptr, (float*)d_out);
}
